// FASTKAGCN_6640019439798
// MI455X (gfx1250) — matrix operand order repaired and checked by reading, not compiled, not run on hardware
//
#include <hip/hip_runtime.h>

// ---------------------------------------------------------------------------
// FastKAN-GCN forward for MI455X (gfx1250, wave32, WMMA).
// Dense FKAN transform runs on v_wmma_f32_16x16x32_f16 with f32 accumulation;
// RBF basis + silu are materialized as f16 in LDS (never hit HBM).
// Graph aggregation is edge-parallel with global_atomic_add_f32 (HBM bound).
// ---------------------------------------------------------------------------

typedef _Float16 v8h  __attribute__((ext_vector_type(8)));
typedef _Float16 v16h __attribute__((ext_vector_type(16)));
typedef float    v8f  __attribute__((ext_vector_type(8)));

#define NFEAT   128
#define NHID    128
#define GRIDN   4
#define KSP     512           // NFEAT * GRIDN
#define KBASE   128
#define ASTRIDE 648           // halves; multiple of 8 -> keeps b128 alignment
#define CLS     10
#define NGRAPH  64
#define NLAYER  3
#define LN_EPS  1e-5f
#define INV_DENOM 0.75f       // 1 / ((2-(-2))/(G-1)) = 1/(4/3)

// Build a 16-half WMMA fragment from two aligned 8-half runs.
__device__ __forceinline__ v16h frag_ld(const _Float16* p0, const _Float16* p1) {
  v8h lo = *(const v8h*)p0;
  v8h hi = *(const v8h*)p1;
  v16h r;
#pragma unroll
  for (int i = 0; i < 8; ++i) { r[i] = lo[i]; r[i + 8] = hi[i]; }
  return r;
}

__device__ __forceinline__ v8f wmma_f16(v16h a, v16h b, v8f c) {
  return __builtin_amdgcn_wmma_f32_16x16x32_f16(false, a, false, b,
                                                (short)0, c, false, false);
}

// --------------------------- utility kernels -------------------------------

__global__ void zero_f32(float* __restrict__ p, long long n) {
  long long i = (long long)blockIdx.x * blockDim.x + threadIdx.x;
  if (i < n) p[i] = 0.f;
}

__global__ void wconv_kernel(const float* __restrict__ sW,
                             const float* __restrict__ bW,
                             _Float16* __restrict__ w16, int tot) {
  int i = blockIdx.x * blockDim.x + threadIdx.x;
  if (i >= tot) return;
  const int PER = NHID * KSP + NHID * KBASE;   // 81920 per layer
  int l = i / PER, r = i % PER;
  float v = (r < NHID * KSP)
              ? sW[(size_t)l * NHID * KSP + r]
              : bW[(size_t)l * NHID * KBASE + (r - NHID * KSP)];
  w16[i] = (_Float16)v;
}

__global__ void deg_kernel(const int* __restrict__ ei, float* __restrict__ deg,
                           int E, int n) {
  long long i = (long long)blockIdx.x * blockDim.x + threadIdx.x;
  long long total = (long long)E + n;
  if (i >= total) return;
  int d = (i < E) ? ei[E + i] : (int)(i - E);   // self-loop tail
  atomicAdd(&deg[d], 1.f);
}

__global__ void dinv_kernel(float* __restrict__ deg, int n) {
  int i = blockIdx.x * blockDim.x + threadIdx.x;
  if (i >= n) return;
  float d = deg[i];
  deg[i] = (d > 0.f) ? rsqrtf(d) : 0.f;
}

// --------------------------- FKAN WMMA kernel ------------------------------
// Block = 128 threads (4 waves), 16 node rows.  LDS A panel [16][640] f16:
// cols 0..511 = RBF basis of layernorm(x); cols 512..639 = silu(x).
// Wave w computes output columns [32w, 32w+32) as two 16x16 f32 tiles.

__global__ __launch_bounds__(128)
void fkan_wmma_kernel(const float* __restrict__ xin,
                      const float* __restrict__ ln_s,
                      const float* __restrict__ ln_b,
                      const _Float16* __restrict__ sw16,  // [128][512] f16
                      const float* __restrict__ sb,
                      const _Float16* __restrict__ bw16,  // [128][128] f16
                      const float* __restrict__ bb,
                      float* __restrict__ h, int n) {
  __shared__ __align__(16) _Float16 aT[16 * ASTRIDE];
  __shared__ float xs[16 * NFEAT];
  __shared__ float stats[32];

  const int tid = threadIdx.x;
  const int node0 = blockIdx.x * 16;

  // ---- stage x tile ----
  for (int i = tid; i < 16 * NFEAT; i += 128) {
    int r = i >> 7, c = i & 127;
    int node = node0 + r;
    xs[i] = (node < n) ? xin[(size_t)node * NFEAT + c] : 0.f;
  }
  __syncthreads();

  // ---- per-row layernorm stats ----
  if (tid < 16) {
    float m = 0.f;
    for (int c = 0; c < NFEAT; ++c) m += xs[tid * NFEAT + c];
    m *= (1.f / NFEAT);
    float s = 0.f;
    for (int c = 0; c < NFEAT; ++c) {
      float d = xs[tid * NFEAT + c] - m;
      s += d * d;
    }
    s *= (1.f / NFEAT);
    stats[tid * 2]     = m;
    stats[tid * 2 + 1] = rsqrtf(s + LN_EPS);
  }
  __syncthreads();

  // ---- build f16 A panel: RBF basis + silu ----
  for (int i = tid; i < 16 * NFEAT; i += 128) {
    int r = i >> 7, c = i & 127;
    float v = xs[r * NFEAT + c];
    float z = (v - stats[r * 2]) * stats[r * 2 + 1] * ln_s[c] + ln_b[c];
#pragma unroll
    for (int g = 0; g < GRIDN; ++g) {
      float gv = -2.f + (4.f / 3.f) * (float)g;
      float t  = (z - gv) * INV_DENOM;
      aT[r * ASTRIDE + c * GRIDN + g] = (_Float16)__expf(-t * t);
    }
    aT[r * ASTRIDE + KSP + c] = (_Float16)(v / (1.f + __expf(-v)));
  }
  __syncthreads();

  // ---- WMMA phase ----
  const int wave = tid >> 5;
  const int lane = tid & 31;
  const int rsub = lane & 15;
  const int kbA  = (lane >> 4) * 16;   // A: K = {ks..ks+15} contiguous
  const int kbB  = (lane >> 4) * 16;   // B: K = {ks..ks+15} contiguous
  const int col0 = wave * 32;

  v8f acc0 = {};
  v8f acc1 = {};
  const _Float16* arow = &aT[rsub * ASTRIDE];
  const _Float16* w0s  = &sw16[(size_t)(col0 + rsub) * KSP];
  const _Float16* w1s  = &sw16[(size_t)(col0 + 16 + rsub) * KSP];

#pragma unroll 4
  for (int kc = 0; kc < KSP; kc += 32) {
    v16h a  = frag_ld(arow + kc + kbA, arow + kc + kbA + 8);
    v16h b0 = frag_ld(w0s + kc + kbB, w0s + kc + kbB + 8);
    v16h b1 = frag_ld(w1s + kc + kbB, w1s + kc + kbB + 8);
    acc0 = wmma_f16(a, b0, acc0);
    acc1 = wmma_f16(a, b1, acc1);
  }

  const _Float16* w0b = &bw16[(size_t)(col0 + rsub) * KBASE];
  const _Float16* w1b = &bw16[(size_t)(col0 + 16 + rsub) * KBASE];
#pragma unroll
  for (int kc = 0; kc < KBASE; kc += 32) {
    v16h a  = frag_ld(arow + KSP + kc + kbA, arow + KSP + kc + kbA + 8);
    v16h b0 = frag_ld(w0b + kc + kbB, w0b + kc + kbB + 8);
    v16h b1 = frag_ld(w1b + kc + kbB, w1b + kc + kbB + 8);
    acc0 = wmma_f16(a, b0, acc0);
    acc1 = wmma_f16(a, b1, acc1);
  }

  // ---- store D tiles (+ spline & base bias) ----
  const int mb = (lane >> 4) * 8;
  const int nn = lane & 15;
  const float bias0 = sb[col0 + nn] + bb[col0 + nn];
  const float bias1 = sb[col0 + 16 + nn] + bb[col0 + 16 + nn];
#pragma unroll
  for (int r2 = 0; r2 < 8; ++r2) {
    int node = node0 + mb + r2;
    if (node < n) {
      h[(size_t)node * NHID + col0 + nn]      = acc0[r2] + bias0;
      h[(size_t)node * NHID + col0 + 16 + nn] = acc1[r2] + bias1;
    }
  }
}

// --------------------------- graph aggregation -----------------------------
// One wave32 per edge; lane handles 4 of 128 features (float4 + f32 atomics).

__global__ void scatter_kernel(const float* __restrict__ h,
                               const int* __restrict__ ei,
                               const float* __restrict__ dinv,
                               float* __restrict__ agg, int E, int n) {
  long long wid = (long long)blockIdx.x * (blockDim.x >> 5) + (threadIdx.x >> 5);
  int lane = threadIdx.x & 31;
  long long total = (long long)E + n;
  if (wid >= total) return;
  int s, d;
  if (wid < E) { s = ei[wid]; d = ei[E + wid]; }
  else         { s = d = (int)(wid - E); }
  float nrm = dinv[s] * dinv[d];
  const float4 v = *(const float4*)&h[(size_t)s * NHID + lane * 4];
  float* ap = &agg[(size_t)d * NHID + lane * 4];
  atomicAdd(ap + 0, nrm * v.x);
  atomicAdd(ap + 1, nrm * v.y);
  atomicAdd(ap + 2, nrm * v.z);
  atomicAdd(ap + 3, nrm * v.w);
}

__global__ void bias_silu_kernel(float* __restrict__ agg,
                                 const float* __restrict__ gcn_b,
                                 long long total) {
  long long i = (long long)blockIdx.x * blockDim.x + threadIdx.x;
  if (i >= total) return;
  float v = agg[i] + gcn_b[i & 127];
  agg[i] = v / (1.f + __expf(-v));
}

__global__ void pool_kernel(const float* __restrict__ x,
                            const int* __restrict__ batch,
                            float* __restrict__ pooled,
                            float* __restrict__ cnt, int n) {
  long long wid = (long long)blockIdx.x * (blockDim.x >> 5) + (threadIdx.x >> 5);
  int lane = threadIdx.x & 31;
  if (wid >= n) return;
  int b = batch[wid];
  const float4 v = *(const float4*)&x[(size_t)wid * NHID + lane * 4];
  float* pp = &pooled[b * NHID + lane * 4];
  atomicAdd(pp + 0, v.x);
  atomicAdd(pp + 1, v.y);
  atomicAdd(pp + 2, v.z);
  atomicAdd(pp + 3, v.w);
  if (lane == 0) atomicAdd(&cnt[b], 1.f);
}

// --------------------------- readout (64x10, tiny) -------------------------

__global__ __launch_bounds__(128)
void readout_kernel(const float* __restrict__ pooled,
                    const float* __restrict__ cnt,
                    const float* __restrict__ ln_s, const float* __restrict__ ln_b,
                    const float* __restrict__ sW, const float* __restrict__ sb,
                    const float* __restrict__ bW, const float* __restrict__ bb,
                    float* __restrict__ out) {
  __shared__ float xrow[NHID];
  __shared__ float z[NHID];
  __shared__ float sl[NHID];
  __shared__ float mv[2];
  __shared__ float o[CLS];

  int g = blockIdx.x, t = threadIdx.x;
  float inv = 1.f / fmaxf(cnt[g], 1.f);
  xrow[t] = pooled[g * NHID + t] * inv;
  __syncthreads();
  if (t == 0) {
    float m = 0.f;
    for (int i = 0; i < NHID; ++i) m += xrow[i];
    m *= (1.f / NHID);
    float s = 0.f;
    for (int i = 0; i < NHID; ++i) { float d = xrow[i] - m; s += d * d; }
    s *= (1.f / NHID);
    mv[0] = m; mv[1] = rsqrtf(s + LN_EPS);
  }
  __syncthreads();
  z[t]  = (xrow[t] - mv[0]) * mv[1] * ln_s[t] + ln_b[t];
  sl[t] = xrow[t] / (1.f + __expf(-xrow[t]));
  __syncthreads();
  if (t < CLS) {
    float acc = sb[t] + bb[t];
    for (int f = 0; f < NHID; ++f) {
      float zz = z[f];
#pragma unroll
      for (int gg = 0; gg < GRIDN; ++gg) {
        float gv = -2.f + (4.f / 3.f) * (float)gg;
        float d  = (zz - gv) * INV_DENOM;
        acc += __expf(-d * d) * sW[t * KSP + f * GRIDN + gg];
      }
      acc += sl[f] * bW[t * NHID + f];
    }
    o[t] = acc;
  }
  __syncthreads();
  if (t == 0) {
    float mx = o[0];
    for (int c = 1; c < CLS; ++c) mx = fmaxf(mx, o[c]);
    float se = 0.f;
    for (int c = 0; c < CLS; ++c) se += __expf(o[c] - mx);
    float lse = mx + logf(se);
    for (int c = 0; c < CLS; ++c) out[g * CLS + c] = o[c] - lse;
  }
}

// --------------------------- host orchestration ----------------------------

extern "C" void kernel_launch(void* const* d_in, const int* in_sizes, int n_in,
                              void* d_out, int out_size, void* d_ws, size_t ws_size,
                              hipStream_t stream) {
  (void)n_in; (void)out_size; (void)ws_size;
  const float* x     = (const float*)d_in[0];
  const int*   ei    = (const int*)d_in[1];
  const int*   batch = (const int*)d_in[2];
  const float* ln_s  = (const float*)d_in[3];
  const float* ln_b  = (const float*)d_in[4];
  const float* sW    = (const float*)d_in[5];
  const float* sb    = (const float*)d_in[6];
  const float* bW    = (const float*)d_in[7];
  const float* bb    = (const float*)d_in[8];
  const float* gcn_b = (const float*)d_in[9];
  const float* ro_ln_s = (const float*)d_in[10];
  const float* ro_ln_b = (const float*)d_in[11];
  const float* ro_sW   = (const float*)d_in[12];
  const float* ro_sb   = (const float*)d_in[13];
  const float* ro_bW   = (const float*)d_in[14];
  const float* ro_bb   = (const float*)d_in[15];

  const int N = in_sizes[0] / NFEAT;
  const int E = in_sizes[1] / 2;

  // workspace carve-out (256B aligned slabs)
  char* p = (char*)d_ws;
  auto carve = [&](size_t bytes) -> char* {
    char* r = p;
    p += (bytes + 255) & ~(size_t)255;
    return r;
  };
  float*    deg    = (float*)carve((size_t)N * 4);               // deg -> dinv
  float*    bufP   = (float*)carve((size_t)N * NFEAT * 4);       // h
  float*    bufQ   = (float*)carve((size_t)N * NFEAT * 4);       // agg / x
  float*    pooled = (float*)carve((size_t)NGRAPH * NFEAT * 4);
  float*    cnt    = (float*)carve((size_t)NGRAPH * 4);
  _Float16* w16    = (_Float16*)carve((size_t)NLAYER * (NHID * KSP + NHID * KBASE) * 2);

  // one-time f32 -> f16 weight conversion (0.5 MB, lives in L2)
  {
    int tot = NLAYER * (NHID * KSP + NHID * KBASE);
    wconv_kernel<<<(tot + 255) / 256, 256, 0, stream>>>(sW, bW, w16, tot);
  }

  // GCN degree normalization (with self-loops)
  zero_f32<<<(N + 255) / 256, 256, 0, stream>>>(deg, N);
  {
    long long tot = (long long)E + N;
    deg_kernel<<<(int)((tot + 255) / 256), 256, 0, stream>>>(ei, deg, E, N);
  }
  dinv_kernel<<<(N + 255) / 256, 256, 0, stream>>>(deg, N);

  const float* xin = x;
  for (int l = 0; l < NLAYER; ++l) {
    const _Float16* sw16 = w16 + (size_t)l * (NHID * KSP + NHID * KBASE);
    const _Float16* bw16 = sw16 + (size_t)NHID * KSP;

    fkan_wmma_kernel<<<(N + 15) / 16, 128, 0, stream>>>(
        xin, ln_s + l * NFEAT, ln_b + l * NFEAT,
        sw16, sb + l * NHID, bw16, bb + l * NHID, bufP, N);

    long long tot = (long long)N * NFEAT;
    zero_f32<<<(int)((tot + 255) / 256), 256, 0, stream>>>(bufQ, tot);

    long long waves = (long long)E + N;
    scatter_kernel<<<(int)((waves + 7) / 8), 256, 0, stream>>>(bufP, ei, deg, bufQ, E, N);

    bias_silu_kernel<<<(int)((tot + 255) / 256), 256, 0, stream>>>(bufQ, gcn_b + l * NHID, tot);
    xin = bufQ;
  }

  // mean pool per graph
  zero_f32<<<(NGRAPH * NFEAT + 255) / 256, 256, 0, stream>>>(pooled, NGRAPH * NFEAT);
  zero_f32<<<1, 256, 0, stream>>>(cnt, NGRAPH);
  pool_kernel<<<(N + 7) / 8, 256, 0, stream>>>(bufQ, batch, pooled, cnt, N);

  // readout FKAN + log_softmax
  readout_kernel<<<NGRAPH, 128, 0, stream>>>(pooled, cnt, ro_ln_s, ro_ln_b,
                                             ro_sW, ro_sb, ro_bW, ro_bb,
                                             (float*)d_out);
}
